// PMBlock_5291399709421
// MI455X (gfx1250) — matrix operand order repaired and checked by reading, not compiled, not run on hardware
//
#include <hip/hip_runtime.h>
#include <hip/hip_bf16.h>

// ---------------------------------------------------------------------------
// Hyperbolic/Euclidean MoE transformer block for MI455X (gfx1250).
// Compute-bound (~120 GFLOP vs ~60MB traffic @ 23.3 TB/s): all GEMMs and
// attention contractions run on v_wmma_f32_16x16x32_bf16 (fp32 accumulate);
// GEMM tiles are streamed global->LDS with the CDNA5 async copy engine
// (ASYNCcnt, double buffered); manifold math stays in fp32 VALU.
// ---------------------------------------------------------------------------

#define S_   512
#define B_   16
#define E_   512
#define H_   8
#define D_   64
#define F_   2048
#define NT_  8192            // S_*B_ tokens
#define BH_  128             // B_*H_
#define EPS_   1e-6f
#define MAXT_  (1.0f - 1e-7f)
#define PROJM_ (1.0f - 1e-5f)

typedef __bf16 bf16;
typedef __bf16 bf16x16 __attribute__((ext_vector_type(16)));
typedef float  f32x8   __attribute__((ext_vector_type(8)));
typedef unsigned int u32x4 __attribute__((ext_vector_type(4)));

union FragU { bf16x16 v; u32x4 q[2]; };

__device__ __forceinline__ f32x8 zero8() { f32x8 z = {0,0,0,0,0,0,0,0}; return z; }

__device__ __forceinline__ f32x8 wmma_bf16(bf16x16 a, bf16x16 b, f32x8 c) {
  return __builtin_amdgcn_wmma_f32_16x16x32_bf16(false, a, false, b, (short)0, c, false, false);
}

// Per-lane async global->LDS 16B copy (CDNA5 async engine, ASYNCcnt-tracked).
__device__ __forceinline__ void async_copy16(unsigned lds_off, const void* gaddr) {
  asm volatile("global_load_async_to_lds_b128 %0, %1, off"
               :: "v"(lds_off), "v"(gaddr) : "memory");
}

// A-fragment (16x32 MxK) from row-major [rows x ld] bf16.
// lane<16: row=lane,   K chunks [0..7] and [16..23]
// lane>=16: row=lane-16, K chunks [8..15] and [24..31]
__device__ __forceinline__ bf16x16 load_fragA(const bf16* base, int ld, int row0, int k0) {
  int lane = threadIdx.x & 31;
  const bf16* p = base + (size_t)(row0 + (lane & 15)) * ld + k0 + ((lane >> 4) << 3);
  FragU u;
  u.q[0] = *reinterpret_cast<const u32x4*>(p);
  u.q[1] = *reinterpret_cast<const u32x4*>(p + 16);
  return u.v;
}
// B-fragment (32x16 KxN) from row-major [N x ld] bf16 holding B^T rows.
// lane<16: col n=lane,    K chunks [0..7] and [16..23]
// lane>=16: col n=lane-16, K chunks [8..15] and [24..31]
__device__ __forceinline__ bf16x16 load_fragB(const bf16* base, int ld, int n0, int k0) {
  int lane = threadIdx.x & 31;
  const bf16* p = base + (size_t)(n0 + (lane & 15)) * ld + k0 + ((lane >> 4) << 3);
  FragU u;
  u.q[0] = *reinterpret_cast<const u32x4*>(p);
  u.q[1] = *reinterpret_cast<const u32x4*>(p + 16);
  return u.v;
}

// wave32 reductions
__device__ __forceinline__ float rsum16(float v) {
#pragma unroll
  for (int o = 1; o < 16; o <<= 1) v += __shfl_xor(v, o, 32);
  return v;
}
__device__ __forceinline__ float rmax16(float v) {
#pragma unroll
  for (int o = 1; o < 16; o <<= 1) v = fmaxf(v, __shfl_xor(v, o, 32));
  return v;
}
__device__ __forceinline__ float rsum32(float v) {
#pragma unroll
  for (int o = 1; o < 32; o <<= 1) v += __shfl_xor(v, o, 32);
  return v;
}
__device__ __forceinline__ float blk_sum256(float v, float* sh) {
  v = rsum32(v);
  if ((threadIdx.x & 31) == 0) sh[threadIdx.x >> 5] = v;
  __syncthreads();
  float r = 0.f;
#pragma unroll
  for (int i = 0; i < 8; i++) r += sh[i];
  __syncthreads();
  return r;
}
__device__ __forceinline__ float artanh_f(float x) {
  x = fminf(x, MAXT_);
  return 0.5f * __logf((1.f + x) / (1.f - x));
}

// ---------------------------------------------------------------------------
// Generic TN GEMM: C = A[MxK] * W[NxK]^T (+bias)(+relu)(+res). bf16 in, f32 acc.
// Double-buffered async global->LDS staging (3x b128 per thread per K-tile),
// fragments re-read from LDS via ds_load_b128. flags: 1=relu, 2=residual.
// ---------------------------------------------------------------------------
__global__ __launch_bounds__(256) void k_gemm(
    const bf16* __restrict__ A, const bf16* __restrict__ W,
    const float* __restrict__ bias, const float* __restrict__ res,
    float* __restrict__ Cf, bf16* __restrict__ Cb,
    int M, int N, int K, int flags) {
  __shared__ bf16 sA[2][64][32];    // 2 x 4KB
  __shared__ bf16 sW[2][128][32];   // 2 x 8KB
  int tid = threadIdx.x;
  int w = tid >> 5;
  int lane = tid & 31;
  int mrow = (w >> 2) * 32;              // wave row tile within block (0/32)
  int ncol = (w & 3) * 32;               // wave col tile within block (0..96)
  int rowBase = blockIdx.y * 64;
  int colBase = blockIdx.x * 128;

  // Per-thread staging slots: A = 1 chunk, W = 2 chunks (16B each).
  int ar = tid >> 2, ac = (tid & 3) * 8;
  int wr0 = tid >> 2, wc0 = (tid & 3) * 8;
  int wr1 = (tid + 256) >> 2, wc1 = ((tid + 256) & 3) * 8;

  auto stage = [&](int kk, int buf) {
    async_copy16((unsigned)(size_t)&sA[buf][ar][ac],
                 A + (size_t)(rowBase + ar) * K + kk + ac);
    async_copy16((unsigned)(size_t)&sW[buf][wr0][wc0],
                 W + (size_t)(colBase + wr0) * K + kk + wc0);
    async_copy16((unsigned)(size_t)&sW[buf][wr1][wc1],
                 W + (size_t)(colBase + wr1) * K + kk + wc1);
  };

  f32x8 acc00 = zero8(), acc01 = zero8(), acc10 = zero8(), acc11 = zero8();
  const int nk = K >> 5;
  stage(0, 0);
  for (int kt = 0; kt < nk; kt++) {
    int buf = kt & 1;
    if (kt + 1 < nk) {
      stage((kt + 1) * 32, buf ^ 1);
      // newest 3 outstanding belong to the next stage; oldest 3 (this buf) done
      asm volatile("s_wait_asynccnt 0x3" ::: "memory");
    } else {
      asm volatile("s_wait_asynccnt 0x0" ::: "memory");
    }
    __syncthreads();
    bf16x16 a0 = load_fragA(&sA[buf][0][0], 32, mrow, 0);
    bf16x16 a1 = load_fragA(&sA[buf][0][0], 32, mrow + 16, 0);
    bf16x16 b0 = load_fragB(&sW[buf][0][0], 32, ncol, 0);
    bf16x16 b1 = load_fragB(&sW[buf][0][0], 32, ncol + 16, 0);
    acc00 = wmma_bf16(a0, b0, acc00);
    acc01 = wmma_bf16(a0, b1, acc01);
    acc10 = wmma_bf16(a1, b0, acc10);
    acc11 = wmma_bf16(a1, b1, acc11);
    __syncthreads();
  }
  int hf = lane >> 4, r16 = lane & 15;
#pragma unroll
  for (int i = 0; i < 2; i++) {
#pragma unroll
    for (int j = 0; j < 2; j++) {
      f32x8 a = (i == 0) ? ((j == 0) ? acc00 : acc01) : ((j == 0) ? acc10 : acc11);
      int c = colBase + ncol + j * 16 + r16;
      float bv = bias ? bias[c] : 0.f;
#pragma unroll
      for (int r = 0; r < 8; r++) {
        size_t idx = (size_t)(rowBase + mrow + i * 16 + r + 8 * hf) * N + c;
        float v = a[r] + bv;
        if (flags & 1) v = fmaxf(v, 0.f);
        if (flags & 2) v += res[idx];
        if (Cf) Cf[idx] = v;
        if (Cb) Cb[idx] = (bf16)v;
      }
    }
  }
}

__global__ void k_cast_bf16(const float* __restrict__ in, bf16* __restrict__ out, int n) {
  int i = blockIdx.x * 256 + threadIdx.x;
  if (i < n) out[i] = (bf16)in[i];
}

// ---------------------------------------------------------------------------
// Euclid layernorm -> bf16
// ---------------------------------------------------------------------------
__global__ __launch_bounds__(256) void k_layernorm(
    const float* __restrict__ x, const float* __restrict__ g, const float* __restrict__ bb,
    bf16* __restrict__ out) {
  __shared__ float sh[8];
  int row = blockIdx.x, tid = threadIdx.x;
  const float* xr = x + (size_t)row * E_;
  float v0 = xr[tid], v1 = xr[tid + 256];
  float mu = blk_sum256(v0 + v1, sh) * (1.f / E_);
  float d0 = v0 - mu, d1 = v1 - mu;
  float var = blk_sum256(d0 * d0 + d1 * d1, sh) * (1.f / E_);
  float is = rsqrtf(var + 1e-5f);
  bf16* orow = out + (size_t)row * E_;
  orow[tid]       = (bf16)(d0 * is * g[tid] + bb[tid]);
  orow[tid + 256] = (bf16)(d1 * is * g[tid + 256] + bb[tid + 256]);
}

// Hyperbolic: expmap0(LN(logmap0(x))), optional final projx; outputs bf16 + row norm
__global__ __launch_bounds__(256) void k_hyp_ln(
    const float* __restrict__ x, const float* __restrict__ g, const float* __restrict__ bb,
    bf16* __restrict__ out, float* __restrict__ nOut, int do_projx) {
  __shared__ float sh[8];
  int row = blockIdx.x, tid = threadIdx.x;
  const float* xr = x + (size_t)row * E_;
  float v0 = xr[tid], v1 = xr[tid + 256];
  float n2 = blk_sum256(v0 * v0 + v1 * v1, sh);
  float n = sqrtf(n2 + 1e-15f);
  float a = artanh_f(n) / n;
  float y0 = a * v0, y1 = a * v1;
  float mu = blk_sum256(y0 + y1, sh) * (1.f / E_);
  float d0 = y0 - mu, d1 = y1 - mu;
  float var = blk_sum256(d0 * d0 + d1 * d1, sh) * (1.f / E_);
  float is = rsqrtf(var + 1e-5f);
  float z0 = d0 * is * g[tid] + bb[tid];
  float z1 = d1 * is * g[tid + 256] + bb[tid + 256];
  float nz2 = blk_sum256(z0 * z0 + z1 * z1, sh);
  float nz = sqrtf(nz2 + 1e-15f);
  float e = tanhf(nz) / nz;
  float w0 = e * z0, w1 = e * z1;
  float rn = tanhf(nz);
  float s = 1.f;
  if (do_projx && rn > PROJM_) s = PROJM_ / rn;
  w0 *= s; w1 *= s;
  float nf2 = blk_sum256(w0 * w0 + w1 * w1, sh);
  bf16* orow = out + (size_t)row * E_;
  orow[tid] = (bf16)w0; orow[tid + 256] = (bf16)w1;
  if (tid == 0) nOut[row] = sqrtf(nf2 + 1e-15f);
}

// ---------------------------------------------------------------------------
// man_linear epilogue: given mx = (projx x)@W^T and nx = |x|, compute
// projx(mobius_add(mobius_matvec_scale(mx,nx), expmap0(b))), then optional
// per-head projx / relu; emits f32/bf16/row-norm as requested.
// flags: 1=per-head projx (dim 512), 2=relu
// ---------------------------------------------------------------------------
template <int EPT>
__global__ __launch_bounds__(256) void k_manlin_post(
    const float* __restrict__ mx, const float* __restrict__ nxA, const float* __restrict__ bias,
    float* __restrict__ outF, bf16* __restrict__ outB, float* __restrict__ nxOut, int flags) {
  const int dim = EPT * 256;
  __shared__ float sh[8];
  __shared__ float shH[H_];
  int row = blockIdx.x, tid = threadIdx.x;
  const float* mr = mx + (size_t)row * dim;
  float v[EPT], bv[EPT];
  float smm = 0, smb = 0, sbb = 0;
#pragma unroll
  for (int i = 0; i < EPT; i++) {
    v[i] = mr[tid + i * 256];
    bv[i] = bias[tid + i * 256];
    smm += v[i] * v[i]; smb += v[i] * bv[i]; sbb += bv[i] * bv[i];
  }
  float nmx2 = blk_sum256(smm, sh);
  float dmb  = blk_sum256(smb, sh);
  float nb2  = blk_sum256(sbb, sh);
  float nmx = sqrtf(nmx2 + 1e-15f);
  float nb  = sqrtf(nb2 + 1e-15f);
  float nx = nxA[row];
  bool ok = nmx > 1e-7f;
  float t = tanhf(nmx / fmaxf(nx, 1e-30f) * artanh_f(nx));
  float sc = ok ? t / nmx : 0.f;      // u = sc*mx, |u| = t
  float x2 = ok ? t * t : 0.f;
  float tb = tanhf(nb);
  float eb = tb / nb;                  // vb_vec = eb*bias, |v| = tb
  float y2 = tb * tb;
  float xy = sc * eb * dmb;
  float Am = 1.f + 2.f * xy + y2;
  float Bm = 1.f - x2;
  float den = fmaxf(1.f + 2.f * xy + x2 * y2, EPS_);
  float w[EPT]; float sww = 0;
#pragma unroll
  for (int i = 0; i < EPT; i++) { w[i] = (Am * sc * v[i] + Bm * eb * bv[i]) / den; sww += w[i] * w[i]; }
  float nw2 = blk_sum256(sww, sh);
  float nw = sqrtf(nw2 + 1e-15f);
  if (nw > PROJM_) {
    float s = PROJM_ / nw;
#pragma unroll
    for (int i = 0; i < EPT; i++) w[i] *= s;
  }
  if (flags & 1) {  // per-head projx (dim==512, heads of 64)
    if (tid < H_) shH[tid] = 0.f;
    __syncthreads();
#pragma unroll
    for (int i = 0; i < EPT; i++) atomicAdd(&shH[(tid + i * 256) >> 6], w[i] * w[i]);
    __syncthreads();
#pragma unroll
    for (int i = 0; i < EPT; i++) {
      float nh = sqrtf(shH[(tid + i * 256) >> 6] + 1e-15f);
      if (nh > PROJM_) w[i] *= PROJM_ / nh;
    }
    __syncthreads();
  }
  if (flags & 2) {
#pragma unroll
    for (int i = 0; i < EPT; i++) w[i] = fmaxf(w[i], 0.f);
  }
  if (nxOut) {
    float s2 = 0;
#pragma unroll
    for (int i = 0; i < EPT; i++) s2 += w[i] * w[i];
    float nf2 = blk_sum256(s2, sh);
    if (tid == 0) nxOut[row] = sqrtf(nf2 + 1e-15f);
  }
#pragma unroll
  for (int i = 0; i < EPT; i++) {
    size_t idx = (size_t)row * dim + tid + i * 256;
    if (outF) outF[idx] = w[i];
    if (outB) outB[idx] = (bf16)w[i];
  }
}

__global__ __launch_bounds__(256) void k_projx_rows(
    const float* __restrict__ x, bf16* __restrict__ out, float* __restrict__ nOut) {
  __shared__ float sh[8];
  int row = blockIdx.x, tid = threadIdx.x;
  const float* xr = x + (size_t)row * E_;
  float v0 = xr[tid], v1 = xr[tid + 256];
  float n2 = blk_sum256(v0 * v0 + v1 * v1, sh);
  float n = sqrtf(n2 + 1e-15f);
  float s = (n > PROJM_) ? PROJM_ / n : 1.f;
  out[(size_t)row * E_ + tid]       = (bf16)(v0 * s);
  out[(size_t)row * E_ + tid + 256] = (bf16)(v1 * s);
  if (tid == 0) nOut[row] = fminf(n, PROJM_);
}

// Mobius gyromidpoint residual: mobius_scalar_mul(0.5,(x1 l1 + x2 l2)/(l1+l2-2))
__global__ __launch_bounds__(256) void k_midpoint(
    const float* __restrict__ x1, const float* __restrict__ x2,
    float* __restrict__ out, int do_projx) {
  __shared__ float sh[8];
  int row = blockIdx.x, tid = threadIdx.x;
  const float* a = x1 + (size_t)row * E_;
  const float* b = x2 + (size_t)row * E_;
  float a0 = a[tid], a1 = a[tid + 256], b0 = b[tid], b1 = b[tid + 256];
  float na = blk_sum256(a0 * a0 + a1 * a1, sh);
  float nbs = blk_sum256(b0 * b0 + b1 * b1, sh);
  float l1 = 2.f / fmaxf(1.f - na, EPS_), l2 = 2.f / fmaxf(1.f - nbs, EPS_);
  float dn = fmaxf(l1 + l2 - 2.f, EPS_);
  float t0 = (a0 * l1 + b0 * l2) / dn, t1 = (a1 * l1 + b1 * l2) / dn;
  float nt2 = blk_sum256(t0 * t0 + t1 * t1, sh);
  float nt = sqrtf(nt2 + 1e-15f);
  float scal = tanhf(0.5f * artanh_f(nt)) / nt;
  float v0 = scal * t0, v1 = scal * t1;
  if (do_projx) {
    float rn = scal * nt;
    if (rn > PROJM_) { float s = PROJM_ / rn; v0 *= s; v1 *= s; }
  }
  out[(size_t)row * E_ + tid] = v0;
  out[(size_t)row * E_ + tid + 256] = v1;
}

// ---------------------------------------------------------------------------
// Attention prep: [N,E] f32 q/k/v -> head-major bf16 qh/kh [BH,S,D],
// vT [BH,D,S] (scaled by lambda for hyp), plus qq/kk (and lambda) rows.
// ---------------------------------------------------------------------------
__global__ __launch_bounds__(256) void k_prep_attn(
    const float* __restrict__ q, const float* __restrict__ k, const float* __restrict__ v,
    bf16* __restrict__ qh, bf16* __restrict__ kh, bf16* __restrict__ vT,
    float* __restrict__ qq, float* __restrict__ kk, float* __restrict__ lamArr) {
  int wv = threadIdx.x >> 5, lane = threadIdx.x & 31;
  int idx = blockIdx.x * 8 + wv;   // bh*S + s
  int bh = idx >> 9;
  int s = idx & 511;
  int b = bh >> 3, h = bh & 7;
  size_t base = (size_t)(s * B_ + b) * E_ + h * D_;
  float q0 = q[base + lane], q1 = q[base + lane + 32];
  float k0 = k[base + lane], k1 = k[base + lane + 32];
  float v0 = v[base + lane], v1 = v[base + lane + 32];
  float sq = rsum32(q0 * q0 + q1 * q1);
  float sk = rsum32(k0 * k0 + k1 * k1);
  size_t hrow = (size_t)bh * S_ + s;
  size_t rbase = hrow * D_;
  qh[rbase + lane] = (bf16)q0; qh[rbase + lane + 32] = (bf16)q1;
  kh[rbase + lane] = (bf16)k0; kh[rbase + lane + 32] = (bf16)k1;
  if (lamArr) {
    float sv = rsum32(v0 * v0 + v1 * v1);
    float lam = 2.f / fmaxf(1.f - sv, EPS_);
    if (lane == 0) lamArr[hrow] = lam;
    v0 *= lam; v1 *= lam;
  }
  vT[((size_t)bh * D_ + lane) * S_ + s] = (bf16)v0;
  vT[((size_t)bh * D_ + lane + 32) * S_ + s] = (bf16)v1;
  if (lane == 0) { qq[hrow] = sq; kk[hrow] = sk; }
}

// ---------------------------------------------------------------------------
// Euclid distance attention (flash-style, WMMA + online softmax).
// Block: 4 waves x 16 queries. Writes bf16 merged output [N,E].
// ---------------------------------------------------------------------------
__global__ __launch_bounds__(128) void k_attn_euclid(
    const bf16* __restrict__ qh, const bf16* __restrict__ kh, const bf16* __restrict__ vT,
    const float* __restrict__ qq, const float* __restrict__ kk,
    bf16* __restrict__ outB) {
  __shared__ bf16 P[4][16][32];
  int wv = threadIdx.x >> 5, lane = threadIdx.x & 31, hf = lane >> 4, r16 = lane & 15;
  int bh = blockIdx.y, b = bh >> 3, h = bh & 7;
  int q0r = blockIdx.x * 64 + wv * 16;
  const bf16* qb = qh + (size_t)bh * S_ * D_;
  const bf16* kb = kh + (size_t)bh * S_ * D_;
  const bf16* vb = vT + (size_t)bh * D_ * S_;
  const float* qqb = qq + (size_t)bh * S_;
  const float* kkb = kk + (size_t)bh * S_;
  bf16x16 qa0 = load_fragA(qb, D_, q0r, 0);
  bf16x16 qa1 = load_fragA(qb, D_, q0r, 32);
  float qqr[8], m[8], lsum[8];
#pragma unroll
  for (int r = 0; r < 8; r++) { qqr[r] = qqb[q0r + r + 8 * hf]; m[r] = -1e30f; lsum[r] = 0.f; }
  f32x8 o0 = zero8(), o1 = zero8(), o2 = zero8(), o3 = zero8();
  for (int jt = 0; jt < S_; jt += 32) {
    if (jt + 32 < S_) {
      __builtin_prefetch(kb + (size_t)(jt + 32 + lane) * D_, 0, 1);
      __builtin_prefetch(vb + (size_t)(lane * 2) * S_ + jt + 32, 0, 1);
    }
    bf16x16 kb0 = load_fragB(kb, D_, jt, 0);
    bf16x16 kb1 = load_fragB(kb, D_, jt, 32);
    bf16x16 kb2 = load_fragB(kb, D_, jt + 16, 0);
    bf16x16 kb3 = load_fragB(kb, D_, jt + 16, 32);
    f32x8 s0 = zero8(), s1 = zero8();
    s0 = wmma_bf16(qa0, kb0, s0); s0 = wmma_bf16(qa1, kb1, s0);
    s1 = wmma_bf16(qa0, kb2, s1); s1 = wmma_bf16(qa1, kb3, s1);
    float kk0 = kkb[jt + r16], kk1 = kkb[jt + 16 + r16];
    float p0[8], p1[8];
#pragma unroll
    for (int r = 0; r < 8; r++) {
      float d20 = fmaxf(qqr[r] + kk0 - 2.f * s0[r], 0.f) + 1e-12f;
      float d21 = fmaxf(qqr[r] + kk1 - 2.f * s1[r], 0.f) + 1e-12f;
      float sc0 = -sqrtf(d20) * 0.125f;
      float sc1 = -sqrtf(d21) * 0.125f;
      float mx = rmax16(fmaxf(sc0, sc1));
      float mn = fmaxf(m[r], mx);
      float al = __expf(m[r] - mn);
      p0[r] = __expf(sc0 - mn); p1[r] = __expf(sc1 - mn);
      float rs = rsum16(p0[r] + p1[r]);
      lsum[r] = lsum[r] * al + rs;
      m[r] = mn;
      o0[r] = o0[r] * al; o1[r] = o1[r] * al; o2[r] = o2[r] * al; o3[r] = o3[r] * al;
    }
#pragma unroll
    for (int r = 0; r < 8; r++) {
      P[wv][r + 8 * hf][r16] = (bf16)p0[r];
      P[wv][r + 8 * hf][16 + r16] = (bf16)p1[r];
    }
    __syncthreads();
    FragU pa;
    {
      const bf16* pr = &P[wv][r16][hf * 8];
      pa.q[0] = *reinterpret_cast<const u32x4*>(pr);
      pa.q[1] = *reinterpret_cast<const u32x4*>(pr + 16);
    }
    o0 = wmma_bf16(pa.v, load_fragB(vb, S_, 0,  jt), o0);
    o1 = wmma_bf16(pa.v, load_fragB(vb, S_, 16, jt), o1);
    o2 = wmma_bf16(pa.v, load_fragB(vb, S_, 32, jt), o2);
    o3 = wmma_bf16(pa.v, load_fragB(vb, S_, 48, jt), o3);
    __syncthreads();
  }
#pragma unroll
  for (int r = 0; r < 8; r++) {
    int s = q0r + r + 8 * hf;
    size_t tb = (size_t)(s * B_ + b) * E_ + h * D_;
    float inv = 1.f / lsum[r];
    outB[tb + r16]      = (bf16)(o0[r] * inv);
    outB[tb + 16 + r16] = (bf16)(o1[r] * inv);
    outB[tb + 32 + r16] = (bf16)(o2[r] * inv);
    outB[tb + 48 + r16] = (bf16)(o3[r] * inv);
  }
}

// ---------------------------------------------------------------------------
// Hyperbolic attention: Poincare distance scores, gyromidpoint aggregation
// (num via WMMA on lambda-scaled V, denom carried per row), then per-head
// mobius_scalar_mul(0.5)+projx. Writes f32 merged output [N,E].
// ---------------------------------------------------------------------------
__global__ __launch_bounds__(128) void k_attn_hyp(
    const bf16* __restrict__ qh, const bf16* __restrict__ kh, const bf16* __restrict__ vT,
    const float* __restrict__ qq, const float* __restrict__ kk, const float* __restrict__ lamA,
    float* __restrict__ outF) {
  __shared__ bf16 P[4][16][32];
  int wv = threadIdx.x >> 5, lane = threadIdx.x & 31, hf = lane >> 4, r16 = lane & 15;
  int bh = blockIdx.y, b = bh >> 3, h = bh & 7;
  int q0r = blockIdx.x * 64 + wv * 16;
  const bf16* qb = qh + (size_t)bh * S_ * D_;
  const bf16* kb = kh + (size_t)bh * S_ * D_;
  const bf16* vb = vT + (size_t)bh * D_ * S_;
  const float* qqb = qq + (size_t)bh * S_;
  const float* kkb = kk + (size_t)bh * S_;
  const float* lmb = lamA + (size_t)bh * S_;
  bf16x16 qa0 = load_fragA(qb, D_, q0r, 0);
  bf16x16 qa1 = load_fragA(qb, D_, q0r, 32);
  float qqr[8], m[8], lsum[8], dnm[8];
#pragma unroll
  for (int r = 0; r < 8; r++) { qqr[r] = qqb[q0r + r + 8 * hf]; m[r] = -1e30f; lsum[r] = 0.f; dnm[r] = 0.f; }
  f32x8 o0 = zero8(), o1 = zero8(), o2 = zero8(), o3 = zero8();
  for (int jt = 0; jt < S_; jt += 32) {
    if (jt + 32 < S_) {
      __builtin_prefetch(kb + (size_t)(jt + 32 + lane) * D_, 0, 1);
      __builtin_prefetch(vb + (size_t)(lane * 2) * S_ + jt + 32, 0, 1);
    }
    bf16x16 kb0 = load_fragB(kb, D_, jt, 0);
    bf16x16 kb1 = load_fragB(kb, D_, jt, 32);
    bf16x16 kb2 = load_fragB(kb, D_, jt + 16, 0);
    bf16x16 kb3 = load_fragB(kb, D_, jt + 16, 32);
    f32x8 s0 = zero8(), s1 = zero8();
    s0 = wmma_bf16(qa0, kb0, s0); s0 = wmma_bf16(qa1, kb1, s0);
    s1 = wmma_bf16(qa0, kb2, s1); s1 = wmma_bf16(qa1, kb3, s1);
    float kk0 = kkb[jt + r16], kk1 = kkb[jt + 16 + r16];
    float la0 = lmb[jt + r16] - 1.f, la1 = lmb[jt + 16 + r16] - 1.f;
    float p0[8], p1[8];
#pragma unroll
    for (int r = 0; r < 8; r++) {
      float qk0 = s0[r], qk1 = s1[r];
      float df0 = fmaxf(qqr[r] + kk0 - 2.f * qk0, 0.f);
      float df1 = fmaxf(qqr[r] + kk1 - 2.f * qk1, 0.f);
      float de0 = fmaxf(1.f - 2.f * qk0 + qqr[r] * kk0, EPS_);
      float de1 = fmaxf(1.f - 2.f * qk1 + qqr[r] * kk1, EPS_);
      float sc0 = -2.f * artanh_f(sqrtf(df0 / de0 + 1e-15f)) * 0.125f;
      float sc1 = -2.f * artanh_f(sqrtf(df1 / de1 + 1e-15f)) * 0.125f;
      float mx = rmax16(fmaxf(sc0, sc1));
      float mn = fmaxf(m[r], mx);
      float al = __expf(m[r] - mn);
      p0[r] = __expf(sc0 - mn); p1[r] = __expf(sc1 - mn);
      float rs = rsum16(p0[r] + p1[r]);
      lsum[r] = lsum[r] * al + rs;
      dnm[r] = dnm[r] * al + p0[r] * la0 + p1[r] * la1;
      m[r] = mn;
      o0[r] = o0[r] * al; o1[r] = o1[r] * al; o2[r] = o2[r] * al; o3[r] = o3[r] * al;
    }
#pragma unroll
    for (int r = 0; r < 8; r++) {
      P[wv][r + 8 * hf][r16] = (bf16)p0[r];
      P[wv][r + 8 * hf][16 + r16] = (bf16)p1[r];
    }
    __syncthreads();
    FragU pa;
    {
      const bf16* pr = &P[wv][r16][hf * 8];
      pa.q[0] = *reinterpret_cast<const u32x4*>(pr);
      pa.q[1] = *reinterpret_cast<const u32x4*>(pr + 16);
    }
    o0 = wmma_bf16(pa.v, load_fragB(vb, S_, 0,  jt), o0);
    o1 = wmma_bf16(pa.v, load_fragB(vb, S_, 16, jt), o1);
    o2 = wmma_bf16(pa.v, load_fragB(vb, S_, 32, jt), o2);
    o3 = wmma_bf16(pa.v, load_fragB(vb, S_, 48, jt), o3);
    __syncthreads();
  }
#pragma unroll
  for (int r = 0; r < 8; r++) {
    int s = q0r + r + 8 * hf;
    size_t tb = (size_t)(s * B_ + b) * E_ + h * D_;
    float inv = 1.f / lsum[r];
    float dT = fmaxf(rsum16(dnm[r]) * inv, EPS_);
    float u0 = o0[r] * inv / dT, u1 = o1[r] * inv / dT;
    float u2 = o2[r] * inv / dT, u3 = o3[r] * inv / dT;
    float n2 = rsum16(u0 * u0 + u1 * u1 + u2 * u2 + u3 * u3);
    float n = sqrtf(n2 + 1e-15f);
    float tt = tanhf(0.5f * artanh_f(n));
    float scl = tt / n;
    if (tt > PROJM_) scl *= PROJM_ / tt;   // projx on the per-head result
    outF[tb + r16]      = u0 * scl;
    outF[tb + 16 + r16] = u1 * scl;
    outF[tb + 32 + r16] = u2 * scl;
    outF[tb + 48 + r16] = u3 * scl;
  }
}

// ---------------------------------------------------------------------------
// Host orchestration
// ---------------------------------------------------------------------------
extern "C" void kernel_launch(void* const* d_in, const int* in_sizes, int n_in,
                              void* d_out, int out_size, void* d_ws, size_t ws_size,
                              hipStream_t stream) {
  (void)in_sizes; (void)n_in; (void)out_size; (void)ws_size;
  const float* X    = (const float*)d_in[0];
  const float* LN1G = (const float*)d_in[1];
  const float* LN1B = (const float*)d_in[2];
  const float* LN2G = (const float*)d_in[3];
  const float* LN2B = (const float*)d_in[4];
  const float* WQ = (const float*)d_in[5];
  const float* BQ = (const float*)d_in[6];
  const float* WK = (const float*)d_in[7];
  const float* BK = (const float*)d_in[8];
  const float* WV = (const float*)d_in[9];
  const float* BV = (const float*)d_in[10];
  const float* WO = (const float*)d_in[11];
  const float* BO = (const float*)d_in[12];
  const float* W1 = (const float*)d_in[13];
  const float* B1 = (const float*)d_in[14];
  const float* W2 = (const float*)d_in[15];
  const float* B2 = (const float*)d_in[16];
  float* OUT = (float*)d_out;

  char* wp = (char*)d_ws;
  auto alloc = [&](size_t bytes) -> void* {
    void* r = (void*)wp;
    wp += (bytes + 255) & ~(size_t)255;
    return r;
  };
  const size_t NE = (size_t)NT_ * E_;
  const size_t NF = (size_t)NT_ * F_;

  bf16* wbQ = (bf16*)alloc((size_t)E_ * E_ * 2);
  bf16* wbK = (bf16*)alloc((size_t)E_ * E_ * 2);
  bf16* wbV = (bf16*)alloc((size_t)E_ * E_ * 2);
  bf16* wbO = (bf16*)alloc((size_t)E_ * E_ * 2);
  bf16* wb1 = (bf16*)alloc((size_t)F_ * E_ * 2);
  bf16* wb2 = (bf16*)alloc((size_t)E_ * F_ * 2);
  bf16* actB = (bf16*)alloc(NE * 2);
  bf16* ffnB = (bf16*)alloc(NF * 2);
  bf16* attB = (bf16*)alloc(NE * 2);
  bf16* qhB = (bf16*)alloc(NE * 2);
  bf16* khB = (bf16*)alloc(NE * 2);
  bf16* vTB = (bf16*)alloc(NE * 2);
  float* f1 = (float*)alloc(NE * 4);
  float* f2 = (float*)alloc(NE * 4);
  float* f3 = (float*)alloc(NE * 4);
  float* f4 = (float*)alloc(NE * 4);
  float* f5 = (float*)alloc(NE * 4);
  float* fBig = (float*)alloc(NF * 4);
  float* qqA = (float*)alloc((size_t)BH_ * S_ * 4);
  float* kkA = (float*)alloc((size_t)BH_ * S_ * 4);
  float* lamA = (float*)alloc((size_t)BH_ * S_ * 4);
  float* nx1 = (float*)alloc((size_t)NT_ * 4);
  float* nx2 = (float*)alloc((size_t)NT_ * 4);

  auto gemm = [&](const bf16* A, const bf16* W, const float* bias, const float* res,
                  float* Cf, bf16* Cb, int M, int N, int K, int flags) {
    dim3 g(N / 128, M / 64);
    k_gemm<<<g, dim3(256), 0, stream>>>(A, W, bias, res, Cf, Cb, M, N, K, flags);
  };
  auto cast = [&](const float* in, bf16* outp, size_t n) {
    k_cast_bf16<<<dim3((unsigned)((n + 255) / 256)), dim3(256), 0, stream>>>(in, outp, (int)n);
  };

  // ------------------ Expert 0: Euclidean ------------------
  {
    const float* x0 = X;
    cast(WQ, wbQ, (size_t)E_ * E_);
    cast(WK, wbK, (size_t)E_ * E_);
    cast(WV, wbV, (size_t)E_ * E_);
    cast(WO, wbO, (size_t)E_ * E_);
    cast(W1, wb1, (size_t)F_ * E_);
    cast(W2, wb2, (size_t)E_ * F_);
    k_layernorm<<<NT_, 256, 0, stream>>>(x0, LN1G, LN1B, actB);
    gemm(actB, wbQ, BQ, nullptr, f2, nullptr, NT_, E_, E_, 0);
    gemm(actB, wbK, BK, nullptr, f3, nullptr, NT_, E_, E_, 0);
    gemm(actB, wbV, BV, nullptr, f4, nullptr, NT_, E_, E_, 0);
    k_prep_attn<<<BH_ * S_ / 8, 256, 0, stream>>>(f2, f3, f4, qhB, khB, vTB, qqA, kkA, nullptr);
    k_attn_euclid<<<dim3(S_ / 64, BH_), 128, 0, stream>>>(qhB, khB, vTB, qqA, kkA, attB);
    gemm(attB, wbO, BO, x0, f5, nullptr, NT_, E_, E_, 2);           // x = o@Wo^T+bo+res
    k_layernorm<<<NT_, 256, 0, stream>>>(f5, LN2G, LN2B, actB);
    gemm(actB, wb1, B1, nullptr, nullptr, ffnB, NT_, F_, E_, 1);    // relu
    gemm(ffnB, wb2, B2, f5, OUT, nullptr, NT_, E_, F_, 3);          // relu + residual
  }

  // ------------------ Expert 1: Hyperbolic ------------------
  {
    const float* x1 = X + NE;
    cast(WQ + (size_t)E_ * E_, wbQ, (size_t)E_ * E_);
    cast(WK + (size_t)E_ * E_, wbK, (size_t)E_ * E_);
    cast(WV + (size_t)E_ * E_, wbV, (size_t)E_ * E_);
    cast(WO + (size_t)E_ * E_, wbO, (size_t)E_ * E_);
    cast(W1 + (size_t)F_ * E_, wb1, (size_t)F_ * E_);
    cast(W2 + (size_t)E_ * F_, wb2, (size_t)E_ * F_);
    const float* bq1 = BQ + E_; const float* bk1 = BK + E_;
    const float* bv1 = BV + E_; const float* bo1 = BO + E_;
    const float* b11 = B1 + F_; const float* b21 = B2 + E_;
    const float* g1 = LN1G + E_; const float* be1 = LN1B + E_;
    const float* g2 = LN2G + E_; const float* be2 = LN2B + E_;

    k_hyp_ln<<<NT_, 256, 0, stream>>>(x1, g1, be1, actB, nx1, 0);
    gemm(actB, wbQ, nullptr, nullptr, f1, nullptr, NT_, E_, E_, 0);
    k_manlin_post<2><<<NT_, 256, 0, stream>>>(f1, nx1, bq1, f2, nullptr, nullptr, 1);
    gemm(actB, wbK, nullptr, nullptr, f1, nullptr, NT_, E_, E_, 0);
    k_manlin_post<2><<<NT_, 256, 0, stream>>>(f1, nx1, bk1, f3, nullptr, nullptr, 1);
    gemm(actB, wbV, nullptr, nullptr, f1, nullptr, NT_, E_, E_, 0);
    k_manlin_post<2><<<NT_, 256, 0, stream>>>(f1, nx1, bv1, f4, nullptr, nullptr, 1);
    k_prep_attn<<<BH_ * S_ / 8, 256, 0, stream>>>(f2, f3, f4, qhB, khB, vTB, qqA, kkA, lamA);
    k_attn_hyp<<<dim3(S_ / 64, BH_), 128, 0, stream>>>(qhB, khB, vTB, qqA, kkA, lamA, f5);
    k_projx_rows<<<NT_, 256, 0, stream>>>(f5, actB, nx2);
    gemm(actB, wbO, nullptr, nullptr, f1, nullptr, NT_, E_, E_, 0);
    k_manlin_post<2><<<NT_, 256, 0, stream>>>(f1, nx2, bo1, f2, nullptr, nullptr, 0);  // xw
    k_midpoint<<<NT_, 256, 0, stream>>>(f2, x1, f3, 0);                                // res
    k_hyp_ln<<<NT_, 256, 0, stream>>>(f3, g2, be2, actB, nx1, 1);
    gemm(actB, wb1, nullptr, nullptr, fBig, nullptr, NT_, F_, E_, 0);
    k_manlin_post<8><<<NT_, 256, 0, stream>>>(fBig, nx1, b11, nullptr, ffnB, nx2, 2);  // relu
    gemm(ffnB, wb2, nullptr, nullptr, f1, nullptr, NT_, E_, F_, 0);
    k_manlin_post<2><<<NT_, 256, 0, stream>>>(f1, nx2, b21, f4, nullptr, nullptr, 2);  // relu
    k_midpoint<<<NT_, 256, 0, stream>>>(f4, f3, OUT + NE, 1);                          // projx
  }
}
